// GCN_markov2_52596169507569
// MI455X (gfx1250) — matrix operand order repaired and checked by reading, not compiled, not run on hardware
//
#include <hip/hip_runtime.h>
#include <hip/hip_bf16.h>
#include <math.h>

typedef __attribute__((ext_vector_type(16))) _Float16 v16h;
typedef __attribute__((ext_vector_type(8)))  _Float16 v8h;
typedef __attribute__((ext_vector_type(8)))  float    v8f;

#define NN 50000
#define EE 300000
#define FIN 128
#define HD 256          // H*D
#define NEG_SLOPE 0.2f

// ---------------------------------------------------------------- utilities
__global__ void k_fill(float* __restrict__ p, float v, int n) {
  for (int i = blockIdx.x * blockDim.x + threadIdx.x; i < n; i += gridDim.x * blockDim.x)
    p[i] = v;
}

__device__ __forceinline__ void atomicMaxFloat(float* addr, float val) {
  // IEEE-754 order-preserving int trick (works for mixed signs, init = -inf)
  if (val >= 0.0f) atomicMax((int*)addr, __float_as_int(val));
  else             atomicMin((unsigned int*)addr, __float_as_uint(val));
}

// -------------------------------------------------- pack W into B-operand layout
// Wpack[((ntile*4 + kstep)*32 + lane)*16 + e] = f16(W[K*256 + col])
//   col = ntile*16 + (lane&15),  K = kstep*32 + (e<8 ? (lane>>4)*8 + e : 16 + (lane>>4)*8 + e-8)
__global__ void k_packW(const float* __restrict__ W, _Float16* __restrict__ Wp) {
  int idx = blockIdx.x * blockDim.x + threadIdx.x;
  if (idx >= 16 * 4 * 32 * 16) return;
  int e     = idx & 15;
  int lane  = (idx >> 4) & 31;
  int kstep = (idx >> 9) & 3;
  int ntile = idx >> 11;
  int col = ntile * 16 + (lane & 15);
  int K   = kstep * 32 + (e < 8 ? ((lane >> 4) << 3) + e : 16 + ((lane >> 4) << 3) + (e - 8));
  Wp[idx] = (_Float16)W[K * HD + col];
}

// -------------------------------------------------- WMMA GEMM: xw = x @ W  [N,256]
__global__ __launch_bounds__(256) void k_gemm(const float* __restrict__ x,
                                              const _Float16* __restrict__ Wp,
                                              float* __restrict__ xw) {
  __shared__ __align__(16) _Float16 sx[16 * FIN];   // 16-row x-tile as f16 (4 KB)
  const int mbase = blockIdx.x * 16;
  const int tid = threadIdx.x;
#pragma unroll
  for (int i = 0; i < 8; ++i) {                      // 2048 elems / 256 threads
    int idx = tid + i * 256;
    int row = idx >> 7, k = idx & 127;
    sx[idx] = (_Float16)x[(size_t)(mbase + row) * FIN + k];
  }
  __syncthreads();

  const int wave = tid >> 5, lane = tid & 31;
  const int ntile0 = wave * 2, ntile1 = ntile0 + 1;
  const int rowA = lane & 15;
  const int kgrp = (lane >> 4) << 3;                 // 0 or 8
  v8f acc0 = {}, acc1 = {};
#pragma unroll
  for (int kstep = 0; kstep < 4; ++kstep) {
    // A operand: 16-bit A 16x32 layout (two aligned 16B LDS reads)
    const _Float16* ap = &sx[rowA * FIN + kstep * 32 + kgrp];
    v8h lo = *(const v8h*)ap;
    v8h hi = *(const v8h*)(ap + 16);
    v16h A = __builtin_shufflevector(lo, hi, 0,1,2,3,4,5,6,7,8,9,10,11,12,13,14,15);
    // B operands: pre-packed, one contiguous 32B load per lane
    v16h B0 = *(const v16h*)(Wp + ((size_t)(ntile0 * 4 + kstep) * 32 + lane) * 16);
    v16h B1 = *(const v16h*)(Wp + ((size_t)(ntile1 * 4 + kstep) * 32 + lane) * 16);
    acc0 = __builtin_amdgcn_wmma_f32_16x16x32_f16(false, A, false, B0, (short)0, acc0, false, false);
    acc1 = __builtin_amdgcn_wmma_f32_16x16x32_f16(false, A, false, B1, (short)0, acc1, false, false);
  }
  // D layout: vgpr r -> M = (lane>=16 ? 8 : 0) + r, N = lane&15
  const int rbase = (lane >> 4) << 3;
  const int c0 = ntile0 * 16 + (lane & 15);
  const int c1 = ntile1 * 16 + (lane & 15);
#pragma unroll
  for (int r = 0; r < 8; ++r) {
    size_t rowoff = (size_t)(mbase + rbase + r) * HD;
    xw[rowoff + c0] = acc0[r];
    xw[rowoff + c1] = acc1[r];
  }
}

// -------------------------------------------------- per-node attention logits
__global__ __launch_bounds__(256) void k_dots(const float* __restrict__ xw,
                                              const float* __restrict__ att_src,
                                              const float* __restrict__ att_dst,
                                              float* __restrict__ a_s,
                                              float* __restrict__ a_d, int n) {
  int warp = (blockIdx.x * blockDim.x + threadIdx.x) >> 5;
  int lane = threadIdx.x & 31;
  if (warp >= n) return;
  const float* xr = xw + (size_t)warp * HD;
  float s0 = 0.f, s1 = 0.f, d0 = 0.f, d1 = 0.f;
#pragma unroll
  for (int i = lane; i < 128; i += 32) {
    float v0 = xr[i], v1 = xr[128 + i];
    s0 += v0 * att_src[i];        s1 += v1 * att_src[128 + i];
    d0 += v0 * att_dst[i];        d1 += v1 * att_dst[128 + i];
  }
#pragma unroll
  for (int off = 16; off; off >>= 1) {
    s0 += __shfl_down(s0, off, 32);
    s1 += __shfl_down(s1, off, 32);
    d0 += __shfl_down(d0, off, 32);
    d1 += __shfl_down(d1, off, 32);
  }
  if (lane == 0) {
    a_s[warp * 2 + 0] = s0;  a_s[warp * 2 + 1] = s1;
    a_d[warp * 2 + 0] = d0;  a_d[warp * 2 + 1] = d1;
  }
}

// -------------------------------------------------- edge pass 1: segment max
__global__ void k_edge_max(const int* __restrict__ ei, const float* __restrict__ a_s,
                           const float* __restrict__ a_d, float* __restrict__ m) {
  int total = (EE + NN) * 2;
  for (int idx = blockIdx.x * blockDim.x + threadIdx.x; idx < total; idx += gridDim.x * blockDim.x) {
    int e = idx >> 1, h = idx & 1;
    int s, t;
    if (e < EE) { s = ei[e]; t = ei[EE + e]; } else { s = t = e - EE; }
    float v = a_s[s * 2 + h] + a_d[t * 2 + h];
    v = v > 0.f ? v : NEG_SLOPE * v;
    atomicMaxFloat(&m[t * 2 + h], v);
  }
}

// -------------------------------------------------- edge pass 2: exp + denom
__global__ void k_edge_exp(const int* __restrict__ ei, const float* __restrict__ a_s,
                           const float* __restrict__ a_d, const float* __restrict__ m,
                           float* __restrict__ ex, float* __restrict__ denom) {
  int total = (EE + NN) * 2;
  for (int idx = blockIdx.x * blockDim.x + threadIdx.x; idx < total; idx += gridDim.x * blockDim.x) {
    int e = idx >> 1, h = idx & 1;
    int s, t;
    if (e < EE) { s = ei[e]; t = ei[EE + e]; } else { s = t = e - EE; }
    float v = a_s[s * 2 + h] + a_d[t * 2 + h];
    v = v > 0.f ? v : NEG_SLOPE * v;
    float xv = expf(v - m[t * 2 + h]);
    ex[idx] = xv;
    atomicAdd(&denom[t * 2 + h], xv);
  }
}

// -------------------------------------------------- edge pass 3: weighted scatter
__global__ __launch_bounds__(256) void k_scatter(const int* __restrict__ ei,
                                                 const float* __restrict__ ex,
                                                 const float* __restrict__ denom,
                                                 const float* __restrict__ xw,
                                                 float* __restrict__ outbuf) {
  int e = blockIdx.x;
  int s, t;
  if (e < EE) { s = ei[e]; t = ei[EE + e]; } else { s = t = e - EE; }
  int tx = threadIdx.x;                 // 0..255  -> h = tx>>7, d = tx&127
  int h = tx >> 7;
  float alpha = ex[e * 2 + h] / denom[t * 2 + h];
  float v = alpha * xw[(size_t)s * HD + tx];
  atomicAdd(&outbuf[(size_t)t * HD + tx], v);
}

// -------------------------------------------------- node head: relu + FC + logsoftmax + pool
__global__ __launch_bounds__(256) void k_head(const float* __restrict__ outbuf,
                                              const float* __restrict__ bias,
                                              const float* __restrict__ fc_w,
                                              const float* __restrict__ fc_b,
                                              const int* __restrict__ batch,
                                              float* __restrict__ pooled) {
  __shared__ float red[8][3];
  int node = blockIdx.x;
  int t = threadIdx.x;
  float z = outbuf[(size_t)node * HD + t] + bias[t];
  z = z > 0.f ? z : 0.f;
  float p0 = z * fc_w[t * 3 + 0];
  float p1 = z * fc_w[t * 3 + 1];
  float p2 = z * fc_w[t * 3 + 2];
#pragma unroll
  for (int off = 16; off; off >>= 1) {
    p0 += __shfl_down(p0, off, 32);
    p1 += __shfl_down(p1, off, 32);
    p2 += __shfl_down(p2, off, 32);
  }
  int lane = t & 31, w = t >> 5;
  if (lane == 0) { red[w][0] = p0; red[w][1] = p1; red[w][2] = p2; }
  __syncthreads();
  if (t == 0) {
    float s0 = fc_b[0], s1 = fc_b[1], s2 = fc_b[2];
#pragma unroll
    for (int i = 0; i < 8; ++i) { s0 += red[i][0]; s1 += red[i][1]; s2 += red[i][2]; }
    float mx = fmaxf(s0, fmaxf(s1, s2));
    float lse = mx + logf(expf(s0 - mx) + expf(s1 - mx) + expf(s2 - mx));
    int b = batch[node];
    atomicAdd(&pooled[b * 3 + 0], s0 - lse);
    atomicAdd(&pooled[b * 3 + 1], s1 - lse);
    atomicAdd(&pooled[b * 3 + 2], s2 - lse);
  }
}

// -------------------------------------------------- final: scale + logsoftmax
__global__ void k_final(const float* __restrict__ pooled, const float* __restrict__ a,
                        float* __restrict__ out) {
  int g = threadIdx.x;
  if (g >= 64) return;
  float av = a[0];
  float l0 = av * pooled[g * 3 + 0];
  float l1 = av * pooled[g * 3 + 1];
  float l2 = av * pooled[g * 3 + 2];
  float mx = fmaxf(l0, fmaxf(l1, l2));
  float lse = mx + logf(expf(l0 - mx) + expf(l1 - mx) + expf(l2 - mx));
  out[g * 3 + 0] = l0 - lse;
  out[g * 3 + 1] = l1 - lse;
  out[g * 3 + 2] = l2 - lse;
}

// ----------------------------------------------------------------- launcher
extern "C" void kernel_launch(void* const* d_in, const int* in_sizes, int n_in,
                              void* d_out, int out_size, void* d_ws, size_t ws_size,
                              hipStream_t stream) {
  const float* x        = (const float*)d_in[0];
  const int*   ei       = (const int*)  d_in[1];
  const int*   batch    = (const int*)  d_in[2];
  const float* W        = (const float*)d_in[3];
  const float* att_src  = (const float*)d_in[4];
  const float* att_dst  = (const float*)d_in[5];
  const float* bias_gat = (const float*)d_in[6];
  const float* fc_w     = (const float*)d_in[7];
  const float* fc_b     = (const float*)d_in[8];
  const float* a        = (const float*)d_in[9];
  float* out = (float*)d_out;

  // workspace carve-up (256B aligned)
  char* ws = (char*)d_ws;
  size_t off = 0;
  auto carve = [&](size_t bytes) -> void* {
    void* p = ws + off;
    off = (off + bytes + 255) & ~(size_t)255;
    return p;
  };
  float*    xw      = (float*)   carve((size_t)NN * HD * 4);        // 51.2 MB
  float*    gat_out = (float*)   carve((size_t)NN * HD * 4);        // 51.2 MB
  float*    a_s     = (float*)   carve((size_t)NN * 2 * 4);
  float*    a_d     = (float*)   carve((size_t)NN * 2 * 4);
  float*    m       = (float*)   carve((size_t)NN * 2 * 4);
  float*    denom   = (float*)   carve((size_t)NN * 2 * 4);
  float*    ex      = (float*)   carve((size_t)(EE + NN) * 2 * 4);  // 2.8 MB
  float*    pooled  = (float*)   carve(64 * 3 * 4);
  _Float16* Wpack   = (_Float16*)carve(16 * 4 * 32 * 16 * 2);       // 64 KB

  // re-init accumulators every call (graph-replay safe)
  k_fill<<<1024, 256, 0, stream>>>(gat_out, 0.f, NN * HD);
  k_fill<<<256, 256, 0, stream>>>(denom, 0.f, NN * 2);
  k_fill<<<256, 256, 0, stream>>>(m, -INFINITY, NN * 2);
  k_fill<<<1, 256, 0, stream>>>(pooled, 0.f, 64 * 3);

  // pack W, WMMA GEMM
  k_packW<<<(16 * 4 * 32 * 16) / 256, 256, 0, stream>>>(W, Wpack);
  k_gemm<<<NN / 16, 256, 0, stream>>>(x, Wpack, xw);

  // attention logits
  k_dots<<<(NN * 32 + 255) / 256, 256, 0, stream>>>(xw, att_src, att_dst, a_s, a_d, NN);

  // segment softmax over incoming edges
  k_edge_max<<<2048, 256, 0, stream>>>(ei, a_s, a_d, m);
  k_edge_exp<<<2048, 256, 0, stream>>>(ei, a_s, a_d, m, ex, denom);

  // weighted message scatter
  k_scatter<<<EE + NN, 256, 0, stream>>>(ei, ex, denom, xw, gat_out);

  // head + pool + final
  k_head<<<NN, 256, 0, stream>>>(gat_out, bias_gat, fc_w, fc_b, batch, pooled);
  k_final<<<1, 64, 0, stream>>>(pooled, a, out);
}
